// BASE_MAMBA_14018773254552
// MI455X (gfx1250) — matrix operand order repaired and checked by reading, not compiled, not run on hardware
//
#include <hip/hip_runtime.h>
#include <hip/hip_bf16.h>

typedef __attribute__((ext_vector_type(16))) _Float16 v16h;
typedef __attribute__((ext_vector_type(8)))  float    v8f;

#define BATCH   4
#define SEQL    1024
#define CIN     20
#define DMODEL  128
#define DSTATE  64
#define DINNER  256
#define DTRANK  9
#define NROWS   (BATCH*SEQL)      // 4096
#define MTILES  (NROWS/16)        // 256
#define MBLOCKS (MTILES/4)        // 64  (4 waves per block, one M-tile each)
#define NLD_DBC 144               // 137 padded to 9 tiles

// ---------- swizzle index helpers (WMMA 16-bit operands in this file's contiguous k order) ----------
// A 16x32 f16: lane = half*16 + (M&15); element e -> K = e + half*16
__device__ __forceinline__ void a_inv(int kk, int row, int& lane, int& e) {
    int half = (kk >> 4) & 1;
    int t = kk - 16 * half;              // in {0..15}
    e = t;
    lane = half * 16 + (row & 15);
}

// ---------- weight prep ----------
// Wc = W_in @ Wp  (512 x 20), written directly in B-swizzled f16, K padded to 32
__global__ void k_comb(const float* __restrict__ W_in, const float* __restrict__ Wp,
                       _Float16* __restrict__ WcSw) {
    int idx = blockIdx.x * blockDim.x + threadIdx.x;     // 32 tiles * 512
    if (idx >= 32 * 512) return;
    int e = idx & 15, lane = (idx >> 4) & 31, nt = idx >> 9;
    int n  = nt * 16 + (lane & 15);                      // < 512
    int kk = ((lane >> 4) << 4) + e;                     // 0..31
    float v = 0.f;
    if (kk < CIN) {
        for (int m = 0; m < DMODEL; ++m) v += W_in[n * DMODEL + m] * Wp[m * CIN + kk];
    }
    WcSw[idx] = (_Float16)v;
}

__global__ void k_bc(const float* __restrict__ W_in, const float* __restrict__ bp,
                     float* __restrict__ bc) {
    int n = blockIdx.x * blockDim.x + threadIdx.x;
    if (n >= 2 * DINNER) return;
    float v = 0.f;
    for (int m = 0; m < DMODEL; ++m) v += W_in[n * DMODEL + m] * bp[m];
    bc[n] = v;
}

// Generic B-matrix swizzle: W (Nrows x Kcols, row-major f32) -> tiled f16
__global__ void k_swzB(const float* __restrict__ W, _Float16* __restrict__ dst,
                       int Nrows, int Kcols, int Ntiles, int Ksteps) {
    int idx = blockIdx.x * blockDim.x + threadIdx.x;
    if (idx >= Ntiles * Ksteps * 512) return;
    int e = idx & 15, lane = (idx >> 4) & 31;
    int unit = idx >> 9;
    int ks = unit % Ksteps, nt = unit / Ksteps;
    int n = nt * 16 + (lane & 15);
    int k = ks * 32 + ((lane >> 4) << 4) + e;
    float v = (n < Nrows && k < Kcols) ? W[(size_t)n * Kcols + k] : 0.f;
    dst[idx] = (_Float16)v;
}

__global__ void k_Aneg(const float* __restrict__ A_log, float* __restrict__ Aneg) {
    int i = blockIdx.x * blockDim.x + threadIdx.x;
    if (i < DINNER * DSTATE) Aneg[i] = -__expf(A_log[i]);
}

// ---------- activation packers ----------
// x (B, CIN, L) f32 -> A-swizzled f16, rows = b*L+l, K padded 20->32
__global__ void k_pack_x(const float* __restrict__ x, _Float16* __restrict__ xA) {
    int idx = blockIdx.x * blockDim.x + threadIdx.x;     // NROWS*32
    if (idx >= NROWS * 32) return;
    int row = idx >> 5, kk = idx & 31;
    int b = row >> 10, l = row & (SEQL - 1);
    float v = (kk < CIN) ? x[((size_t)b * CIN + kk) * SEQL + l] : 0.f;
    int lane, e; a_inv(kk, row, lane, e);
    xA[((size_t)(row >> 4)) * 512 + lane * 16 + e] = (_Float16)v;
}

// generic f32 (NROWS x 256) -> A-swizzled f16, Ksteps=8
__global__ void k_packA(const float* __restrict__ src, _Float16* __restrict__ dst) {
    int idx = blockIdx.x * blockDim.x + threadIdx.x;     // NROWS*256
    if (idx >= NROWS * DINNER) return;
    int row = idx >> 8, k = idx & 255;
    int kk = k & 31, ks = k >> 5;
    int lane, e; a_inv(kk, row, lane, e);
    dst[(((size_t)(row >> 4)) * 8 + ks) * 512 + lane * 16 + e] = (_Float16)src[idx];
}

// ---------- WMMA GEMM ----------
// Block = 128 threads (4 waves). Block owns one N-tile and 4 consecutive M-tiles.
// B panel (Ksteps*512 f16 <= 8KB) is staged to LDS once per block with
// GLOBAL_LOAD_ASYNC_TO_LDS_B128 (ASYNCcnt), then all 4 waves stream it via DS.
__global__ void k_gemm(const _Float16* __restrict__ Asw, const _Float16* __restrict__ Bsw,
                       const float* __restrict__ bias, float* __restrict__ out,
                       int Ntiles, int Ksteps, int N, int Nld) {
    __shared__ __align__(16) _Float16 sB[8 * 512];       // max Ksteps = 8
    int nt = blockIdx.x % Ntiles;
    int mb = blockIdx.x / Ntiles;
    int tid = threadIdx.x;                               // 0..127

    // ---- async stage of B panel into LDS (16B chunks) ----
    const _Float16* bPanel = Bsw + (size_t)nt * Ksteps * 512;
    int nChunks = Ksteps * 64;                           // #16-byte chunks
    for (int i = tid; i < nChunks; i += 128) {
        unsigned ldsOff = (unsigned)(uintptr_t)(&sB[i * 8]);
        unsigned long long ga = (unsigned long long)(uintptr_t)(bPanel + i * 8);
        asm volatile("global_load_async_to_lds_b128 %0, %1, off"
                     :: "v"(ldsOff), "v"(ga) : "memory");
    }
    asm volatile("s_wait_asynccnt 0x0" ::: "memory");
    __syncthreads();

    int wave = tid >> 5, lane = tid & 31;
    int mt = mb * 4 + wave;
    const _Float16* aPtr = Asw + (size_t)mt * Ksteps * 512 + lane * 16;
    const _Float16* bPtr = &sB[lane * 16];
    v8f acc = {};
    for (int ks = 0; ks < Ksteps; ++ks) {
        v16h a = *(const v16h*)aPtr;                     // global_load_b128 x2
        v16h b = *(const v16h*)bPtr;                     // ds_load_b128 x2
        if (ks + 1 < Ksteps) __builtin_prefetch(aPtr + 512, 0, 3);
        acc = __builtin_amdgcn_wmma_f32_16x16x32_f16(false, a, false, b,
                                                     (short)0, acc, false, false);
        aPtr += 512; bPtr += 512;
    }
    int col = nt * 16 + (lane & 15);
    int rowBase = mt * 16 + ((lane >> 4) << 3);
    if (col < N) {
        float bv = bias ? bias[col] : 0.f;
        #pragma unroll
        for (int g = 0; g < 8; ++g)
            out[(size_t)(rowBase + g) * Nld + col] = acc[g] + bv;
    }
}

// ---------- causal depthwise conv + SiLU; emits f32 and A-swizzled f16 ----------
__global__ void k_conv(const float* __restrict__ xz, const float* __restrict__ conv_w,
                       const float* __restrict__ conv_b, float* __restrict__ xc,
                       _Float16* __restrict__ xcA) {
    int idx = blockIdx.x * blockDim.x + threadIdx.x;     // NROWS*256
    if (idx >= NROWS * DINNER) return;
    int row = idx >> 8, d = idx & 255;
    int b = row >> 10, l = row & (SEQL - 1);
    float acc = conv_b[d];
    #pragma unroll
    for (int k = 0; k < 4; ++k) {
        int ls = l + k - 3;
        if (ls >= 0) acc += conv_w[d * 4 + k] * xz[((size_t)(b * SEQL + ls)) * 512 + d];
    }
    float s = acc / (1.f + __expf(-acc));                // silu
    xc[idx] = s;
    int kk = d & 31, ks = d >> 5;
    int lane, e; a_inv(kk, row, lane, e);
    xcA[(((size_t)(row >> 4)) * 8 + ks) * 512 + lane * 16 + e] = (_Float16)s;
}

// ---------- dt = softplus(dt_r @ W_dtT + b_dt) ----------
__global__ void k_dt(const float* __restrict__ dbc, const float* __restrict__ W_dt,
                     const float* __restrict__ b_dt, float* __restrict__ dt) {
    int idx = blockIdx.x * blockDim.x + threadIdx.x;
    if (idx >= NROWS * DINNER) return;
    int row = idx >> 8, d = idx & 255;
    const float* r = dbc + (size_t)row * NLD_DBC;
    float acc = b_dt[d];
    #pragma unroll
    for (int j = 0; j < DTRANK; ++j) acc += r[j] * W_dt[d * DTRANK + j];
    dt[idx] = (acc > 20.f) ? acc : log1pf(__expf(acc));
}

// ---------- selective scan: one wave32 per (b,d), 2 states/lane ----------
__global__ void k_scan(const float* __restrict__ dbc, const float* __restrict__ dt,
                       const float* __restrict__ xc,  const float* __restrict__ xz,
                       const float* __restrict__ Aneg, const float* __restrict__ Dskip,
                       float* __restrict__ yg) {
    int gw   = (blockIdx.x * blockDim.x + threadIdx.x) >> 5;  // 1024 waves
    int lane = threadIdx.x & 31;
    int b = gw >> 8, d = gw & 255;
    float A0 = Aneg[d * DSTATE + lane];
    float A1 = Aneg[d * DSTATE + lane + 32];
    float Dv = Dskip[d];
    float s0 = 0.f, s1 = 0.f;
    size_t rowBase = (size_t)b * SEQL;
    for (int l = 0; l < SEQL; ++l) {
        size_t row = rowBase + l;
        float dtv = dt[row * DINNER + d];
        float xcv = xc[row * DINNER + d];
        const float* dr = dbc + row * NLD_DBC;
        float Bv0 = dr[DTRANK + lane],            Bv1 = dr[DTRANK + 32 + lane];
        float Cv0 = dr[DTRANK + DSTATE + lane],   Cv1 = dr[DTRANK + DSTATE + 32 + lane];
        float dx = dtv * xcv;
        s0 = __expf(dtv * A0) * s0 + dx * Bv0;
        s1 = __expf(dtv * A1) * s1 + dx * Bv1;
        float part = s0 * Cv0 + s1 * Cv1;
        #pragma unroll
        for (int off = 16; off > 0; off >>= 1) part += __shfl_xor(part, off, 32);
        if (lane == 0) {
            float y  = part + xcv * Dv;
            float zv = xz[row * 512 + DINNER + d];
            yg[row * DINNER + d] = y * (zv / (1.f + __expf(-zv)));
        }
    }
}

// ---------- mean pool over L ----------
__global__ void k_pool(const float* __restrict__ outB, float* __restrict__ pooled) {
    int idx = blockIdx.x * blockDim.x + threadIdx.x;     // 4*128
    if (idx >= BATCH * DMODEL) return;
    int b = idx >> 7, m = idx & 127;
    float s = 0.f;
    for (int l = 0; l < SEQL; ++l) s += outB[((size_t)(b * SEQL + l)) * DMODEL + m];
    pooled[idx] = s * (1.f / SEQL);
}

// ---------- MLP head with batch-norm over B ----------
__global__ void k_head(const float* __restrict__ pooled, const float* __restrict__ W1,
                       const float* __restrict__ b1, const float* __restrict__ gamma,
                       const float* __restrict__ beta, const float* __restrict__ W2,
                       const float* __restrict__ b2, float* __restrict__ out) {
    __shared__ float red[BATCH][64];
    int j = threadIdx.x;                                 // 64 threads
    float h[BATCH];
    #pragma unroll
    for (int b = 0; b < BATCH; ++b) h[b] = b1[j];
    for (int k = 0; k < DMODEL; ++k) {
        float w = W1[j * DMODEL + k];
        #pragma unroll
        for (int b = 0; b < BATCH; ++b) h[b] += pooled[b * DMODEL + k] * w;
    }
    float mu = 0.25f * (h[0] + h[1] + h[2] + h[3]);
    float var = 0.f;
    #pragma unroll
    for (int b = 0; b < BATCH; ++b) { float dd = h[b] - mu; var += dd * dd; }
    var *= 0.25f;
    float inv = rsqrtf(var + 1e-5f);
    float w2 = W2[j];
    #pragma unroll
    for (int b = 0; b < BATCH; ++b) {
        float hn = (h[b] - mu) * inv * gamma[j] + beta[j];
        red[b][j] = fmaxf(hn, 0.f) * w2;
    }
    __syncthreads();
    if (j < BATCH) {
        float s = 0.f;
        for (int i = 0; i < 64; ++i) s += red[j][i];
        out[j] = s + b2[0];
    }
}

extern "C" void kernel_launch(void* const* d_in, const int* in_sizes, int n_in,
                              void* d_out, int out_size, void* d_ws, size_t ws_size,
                              hipStream_t stream) {
    const float* x      = (const float*)d_in[0];
    const float* Wp     = (const float*)d_in[1];
    const float* bp     = (const float*)d_in[2];
    const float* W_in   = (const float*)d_in[3];
    const float* conv_w = (const float*)d_in[4];
    const float* conv_b = (const float*)d_in[5];
    const float* W_x    = (const float*)d_in[6];
    const float* W_dt   = (const float*)d_in[7];
    const float* b_dt   = (const float*)d_in[8];
    const float* A_log  = (const float*)d_in[9];
    const float* Dskip  = (const float*)d_in[10];
    const float* W_out  = (const float*)d_in[11];
    const float* W1     = (const float*)d_in[12];
    const float* b1     = (const float*)d_in[13];
    const float* gamma  = (const float*)d_in[14];
    const float* beta   = (const float*)d_in[15];
    const float* W2     = (const float*)d_in[16];
    const float* b2     = (const float*)d_in[17];
    float* out = (float*)d_out;

    char* ws = (char*)d_ws;
    size_t o = 0;
    auto take = [&](size_t bytes) { char* p = ws + o; o += (bytes + 255) & ~(size_t)255; return p; };

    _Float16* xA    = (_Float16*)take((size_t)MTILES * 512 * 2);            // K=32 (1 step)
    _Float16* WcSw  = (_Float16*)take((size_t)32 * 512 * 2);                // 512x32
    float*    bc    = (float*)   take((size_t)512 * 4);
    _Float16* WxSw  = (_Float16*)take((size_t)9 * 8 * 512 * 2);             // 144x256
    _Float16* WoSw  = (_Float16*)take((size_t)8 * 8 * 512 * 2);             // 128x256
    float*    Aneg  = (float*)   take((size_t)DINNER * DSTATE * 4);
    float*    xz    = (float*)   take((size_t)NROWS * 512 * 4);
    float*    xc    = (float*)   take((size_t)NROWS * DINNER * 4);
    _Float16* xcA   = (_Float16*)take((size_t)MTILES * 8 * 512 * 2);
    float*    dbc   = (float*)   take((size_t)NROWS * NLD_DBC * 4);
    float*    dtb   = (float*)   take((size_t)NROWS * DINNER * 4);
    float*    yg    = (float*)   take((size_t)NROWS * DINNER * 4);
    _Float16* ygA   = (_Float16*)take((size_t)MTILES * 8 * 512 * 2);
    float*    outB  = (float*)   take((size_t)NROWS * DMODEL * 4);
    float*    pooled= (float*)   take((size_t)BATCH * DMODEL * 4);
    (void)ws_size; (void)in_sizes; (void)n_in; (void)out_size;

    // weight prep
    k_comb <<<  64, 256, 0, stream>>>(W_in, Wp, WcSw);
    k_bc   <<<   2, 256, 0, stream>>>(W_in, bp, bc);
    k_swzB <<< 144, 256, 0, stream>>>(W_x,   WxSw, 137, 256, 9, 8);
    k_swzB <<< 128, 256, 0, stream>>>(W_out, WoSw, 128, 256, 8, 8);
    k_Aneg <<<  64, 256, 0, stream>>>(A_log, Aneg);

    // xz = x^T @ Wc^T + bc   (fused Wp/W_in projection)
    k_pack_x <<< 512, 256, 0, stream>>>(x, xA);
    k_gemm   <<<32 * MBLOCKS, 128, 0, stream>>>(xA, WcSw, bc, xz, 32, 1, 512, 512);

    // conv + silu
    k_conv <<<4096, 256, 0, stream>>>(xz, conv_w, conv_b, xc, xcA);

    // dbc = xc @ W_x^T
    k_gemm <<<9 * MBLOCKS, 128, 0, stream>>>(xcA, WxSw, nullptr, dbc, 9, 8, 137, NLD_DBC);

    // dt
    k_dt <<<4096, 256, 0, stream>>>(dbc, W_dt, b_dt, dtb);

    // selective scan + gating
    k_scan <<< 128, 256, 0, stream>>>(dbc, dtb, xc, xz, Aneg, Dskip, yg);

    // out = yg @ W_out^T
    k_packA <<<4096, 256, 0, stream>>>(yg, ygA);
    k_gemm  <<<8 * MBLOCKS, 128, 0, stream>>>(ygA, WoSw, nullptr, outB, 8, 8, 128, DMODEL);

    // pool + head
    k_pool <<<   2, 256, 0, stream>>>(outB, pooled);
    k_head <<<   1,  64, 0, stream>>>(pooled, W1, b1, gamma, beta, W2, b2, out);
}
